// GeneralizedAttentionLayer_68092411511013
// MI455X (gfx1250) — matrix operand order repaired and checked by reading, not compiled, not run on hardware
//
#include <hip/hip_runtime.h>

// ---------------------------------------------------------------------------
// GeneralizedAttentionLayer for MI455X (gfx1250, wave32, WMMA bf16)
// D=256, N=2048, B=8.  Fused flash-style attention: att tile never hits HBM.
// Async global->LDS staging (ASYNCcnt), multi-accumulator WMMA ILP,
// hoisted A-fragments, global_prefetch of next n-step.
// ---------------------------------------------------------------------------

typedef __attribute__((ext_vector_type(16))) __bf16 v16bf;
typedef __attribute__((ext_vector_type(8)))  __bf16 v8bf;
typedef __attribute__((ext_vector_type(4)))  __bf16 v4bf;
typedef __attribute__((ext_vector_type(8)))  float  v8f;
typedef __attribute__((ext_vector_type(4)))  int    v4i;

constexpr int kD    = 256;
constexpr int kN    = 2048;
constexpr int kRows = kD + 1;   // 257 rows in Z / output
constexpr int kMT   = 64;       // m-tile (att columns) per block in k2/k3
constexpr int kNT   = 32;       // n-step (att rows streamed) in k2
constexpr int kNT1  = 64;       // n-tile per block in k1

#if defined(__has_builtin)
#  if __has_builtin(__builtin_amdgcn_global_load_async_to_lds_b128)
#    define HAVE_ASYNC_COPY 1
#  endif
#endif

union BFrag { v16bf v; v8bf h[2]; };

// A-matrix fragment (16x32 bf16): lane L holds row M=L%16; K-chunks
// k0 + (L/16)*8 + [0..7]  and  k0+16 + (L/16)*8 + [0..7]   (ISA 7.12.2)
__device__ inline v16bf frag_a(const __bf16* rowbase, int k0, int khalf) {
  BFrag f;
  f.h[0] = *(const v8bf*)(rowbase + k0 + khalf * 8);
  f.h[1] = *(const v8bf*)(rowbase + k0 + 16 + khalf * 8);
  return f.v;
}
// B-matrix fragment (32x16 bf16): lane L holds column N=L%16; K-chunks
// k0 + (L/16)*8 + [0..7]  and  k0+16 + (L/16)*8 + [0..7]   (same as A)
__device__ inline v16bf frag_b(const __bf16* colbase, int k0, int khalf) {
  BFrag f;
  f.h[0] = *(const v8bf*)(colbase + k0 + khalf * 8);
  f.h[1] = *(const v8bf*)(colbase + k0 + 16 + khalf * 8);
  return f.v;
}
__device__ inline v8f wmma_bf16(v16bf a, v16bf b, v8f c) {
  return __builtin_amdgcn_wmma_f32_16x16x32_bf16(
      /*neg_a=*/false, a, /*neg_b=*/false, b,
      /*c_mod=*/(short)0, c, /*reuse_a=*/false, /*reuse_b=*/false);
}

// Per-lane 16-byte memory->LDS copy.  Async (ASYNCcnt-tracked, no VGPR data
// round trip) when the gfx1250 builtin exists; sync vector copy otherwise.
// Builtin signature (from hipcc diagnostic): param0 = addrspace(1) v4i*,
// param1 = addrspace(3) v4i*, then imm offset + imm cpol.
__device__ inline void copy_b128_to_lds(const __bf16* gsrc, __bf16* ldst) {
#if HAVE_ASYNC_COPY
  __builtin_amdgcn_global_load_async_to_lds_b128(
      (__attribute__((address_space(1))) v4i*)gsrc,
      (__attribute__((address_space(3))) v4i*)ldst,
      /*imm offset=*/0, /*cpol=*/0);
#else
  *(v8bf*)ldst = *(const v8bf*)gsrc;
#endif
}
__device__ inline void copy_wait() {
#if HAVE_ASYNC_COPY
#  if __has_builtin(__builtin_amdgcn_s_wait_asynccnt)
  __builtin_amdgcn_s_wait_asynccnt(0);
#  else
  asm volatile("s_wait_asynccnt 0" ::: "memory");
#  endif
#endif
}

// ---------------------------------------------------------------------------
// k0: convert the three 256x256 fp32 weight matrices to bf16 (L2-resident).
// ---------------------------------------------------------------------------
__global__ __launch_bounds__(256) void k0_convert(
    const float* __restrict__ Al, const float* __restrict__ Bl,
    const float* __restrict__ Cl,
    __bf16* __restrict__ Albf, __bf16* __restrict__ Blbf,
    __bf16* __restrict__ Clbf) {
  int i = blockIdx.x * 256 + threadIdx.x;
  if (i < kD * kD) {
    Albf[i] = (__bf16)Al[i];
    Blbf[i] = (__bf16)Bl[i];
    Clbf[i] = (__bf16)Cl[i];
  }
}

// ---------------------------------------------------------------------------
// k1: Ut[b][n][d] = (B_l X)^T, Wt[b][n][d] = (C_l X)^T  (bf16, WMMA),
//     plus Xbf[b][d][n] straight bf16 conversion of X.
// Block = (b, 64-wide n-tile).  X^T tile staged in LDS; the 8 A-fragments
// are loaded once per wave and reused across all 16 output jobs; two
// independent accumulators per k-step hide WMMA latency.
// ---------------------------------------------------------------------------
__global__ __launch_bounds__(256) void k1_uvx(
    const float* __restrict__ Z, const __bf16* __restrict__ Blbf,
    const __bf16* __restrict__ Clbf,
    __bf16* __restrict__ Xbf, __bf16* __restrict__ Ubf,
    __bf16* __restrict__ Wbf) {
  __shared__ __align__(16) __bf16 Xt[kNT1][kD];   // X^T tile [n][j], 32 KB

  const int b   = blockIdx.x / (kN / kNT1);
  const int n0  = (blockIdx.x % (kN / kNT1)) * kNT1;
  const int tid = threadIdx.x;
  const int wv = tid >> 5, lane = tid & 31, lr = lane & 15, lh = lane >> 4;

  // Load + transpose X tile; also emit Xbf[b][d][n] (coalesced bf16).
  {
    const int j = tid;  // 256 threads == 256 rows of X
    const float* src = Z + ((size_t)b * kRows + j) * kN + n0;
    __bf16* xg = Xbf + ((size_t)b * kD + j) * kN + n0;
    for (int c = 0; c < kNT1; c += 4) {
      float4 q = *(const float4*)(src + c);
      __bf16 b0 = (__bf16)q.x, b1 = (__bf16)q.y;
      __bf16 b2 = (__bf16)q.z, b3 = (__bf16)q.w;
      Xt[c + 0][j] = b0; Xt[c + 1][j] = b1;
      Xt[c + 2][j] = b2; Xt[c + 3][j] = b3;
      v4bf p; p[0] = b0; p[1] = b1; p[2] = b2; p[3] = b3;
      *(v4bf*)(xg + c) = p;
    }
  }
  __syncthreads();

  // out(n,d) = sum_j X^T(n,j) * M(d,j) ;  M row d == B-matrix column d.
  const int nsub = wv & 3;            // 4 n-subtiles of 16
  const int grp  = wv >> 2;           // split the 16 d-tiles across 2 groups
  const __bf16* arow = &Xt[nsub * 16 + lr][0];

  v16bf afr[8];                       // hoisted A-fragments (reused 16x)
  #pragma unroll
  for (int k = 0; k < 8; k++) afr[k] = frag_a(arow, k * 32, lh);

  for (int mat = 0; mat < 2; mat++) {
    const __bf16* Wm  = mat ? Clbf : Blbf;
    __bf16*       Out = mat ? Wbf : Ubf;
    for (int tp = 0; tp < 4; tp++) {
      const int dt0 = grp * 8 + tp * 2, dt1 = dt0 + 1;
      const __bf16* bc0 = Wm + (size_t)(dt0 * 16 + lr) * kD;
      const __bf16* bc1 = Wm + (size_t)(dt1 * 16 + lr) * kD;
      v8f a0 = {}, a1 = {};           // two independent WMMA chains
      #pragma unroll
      for (int k = 0; k < 8; k++) {
        a0 = wmma_bf16(afr[k], frag_b(bc0, k * 32, lh), a0);
        a1 = wmma_bf16(afr[k], frag_b(bc1, k * 32, lh), a1);
      }
      // D layout: N(col)=d uses lr; M(row)=n = 8*lh + v
      __bf16* ob = Out + (size_t)b * kN * kD;
      const int nbase = n0 + nsub * 16 + 8 * lh;
      #pragma unroll
      for (int v = 0; v < 8; v++) {
        ob[(size_t)(nbase + v) * kD + dt0 * 16 + lr] = (__bf16)a0[v];
        ob[(size_t)(nbase + v) * kD + dt1 * 16 + lr] = (__bf16)a1[v];
      }
    }
  }
}

// ---------------------------------------------------------------------------
// k2: fused attention.  Block = (b, 64 att-columns m).  Streams n in steps
// of 32: scores (WMMA, split-K dual accumulators) -> relu -> LDS att[m][n]
// (bf16) -> X_att accumulation (WMMA, 64 persistent acc VGPRs/lane) and
// Y_att (VALU).  Staging is async global->LDS (ASYNCcnt) with prefetch of
// the next step.  att never touches HBM (saves ~268 MB of traffic).
// ---------------------------------------------------------------------------
__global__ __launch_bounds__(256) void k2_attn(
    const float* __restrict__ Z, const __bf16* __restrict__ Ubf,
    const __bf16* __restrict__ Wbf, const __bf16* __restrict__ Xbf,
    const float* __restrict__ r_l,
    __bf16* __restrict__ Xatt, float* __restrict__ out) {
  __shared__ __align__(16) __bf16 WB[kMT][kD];   // W columns, 32 KB (block-resident)
  __shared__ __align__(16) __bf16 UA[kNT][kD];   // U rows for this n-step, 16 KB
  __shared__ __align__(16) __bf16 XA[kD][kNT];   // X tile for this n-step, 16 KB
  __shared__ __align__(16) __bf16 ATT[kMT][kNT]; // att tile [m][n], 4 KB
  __shared__ float Yt[kNT];

  const int b   = blockIdx.x / (kN / kMT);
  const int m0  = (blockIdx.x % (kN / kMT)) * kMT;
  const int tid = threadIdx.x;
  const int wv = tid >> 5, lane = tid & 31, lr = lane & 15, lh = lane >> 4;

  // Stage W columns once (async): Wbf[b][m0+m][d] -> WB[m][d]
  {
    const int m = tid >> 2, dc = (tid & 3) * 64;
    const __bf16* s = Wbf + ((size_t)b * kN + m0 + m) * kD + dc;
    #pragma unroll
    for (int i = 0; i < 8; i++)
      copy_b128_to_lds(s + i * 8, &WB[m][dc] + i * 8);
  }

  v8f acc[2][4] = {};            // wave wv owns d in [wv*32, wv*32+32) x 4 m-subtiles
  float yacc = 0.f;
  const float* Yrow = Z + ((size_t)b * kRows + kD) * kN;

  const int ua_n  = tid >> 3, ua_dc = (tid & 7) * 32;      // UA staging map
  const __bf16* ua_src = Ubf + ((size_t)b * kN + ua_n) * kD + ua_dc;
  const __bf16* xa_src = Xbf + ((size_t)b * kD + tid) * kN;

  for (int ns = 0; ns < kN / kNT; ns++) {
    const int n0s = ns * kNT;
    __syncthreads();   // WAR: previous iteration's UA/XA readers done
    // Stage UA: Ubf[b][n][d]   (async, 4x16B per thread)
    #pragma unroll
    for (int i = 0; i < 4; i++)
      copy_b128_to_lds(ua_src + (size_t)n0s * kD + i * 8, &UA[ua_n][ua_dc] + i * 8);
    // Stage XA: Xbf[b][d][n0s..n0s+32)   (async, 4x16B per thread)
    #pragma unroll
    for (int i = 0; i < 4; i++)
      copy_b128_to_lds(xa_src + n0s + i * 8, &XA[tid][0] + i * 8);
    if (tid < kNT) Yt[tid] = Yrow[n0s + tid];
    copy_wait();
    __syncthreads();

    // Prefetch next n-step's staging sources into cache while we compute.
    if (ns + 1 < kN / kNT) {
      __builtin_prefetch(ua_src + (size_t)(n0s + kNT) * kD, 0, 1);
      __builtin_prefetch(xa_src + n0s + kNT, 0, 1);
    }

    // ---- scores: 32x64 tile, 8 waves x one 16x16 tile each, K = 256 ----
    {
      const int nsub = wv >> 2, msub = wv & 3;
      const __bf16* arow = &UA[nsub * 16 + lr][0];
      const __bf16* bcol = &WB[msub * 16 + lr][0];
      v8f s0 = {}, s1 = {};          // split-K: two independent chains
      #pragma unroll
      for (int k = 0; k < kD; k += 64) {
        s0 = wmma_bf16(frag_a(arow, k, lh),      frag_b(bcol, k, lh),      s0);
        s1 = wmma_bf16(frag_a(arow, k + 32, lh), frag_b(bcol, k + 32, lh), s1);
      }
      v8f s = s0 + s1;
      const int mg = m0 + msub * 16 + lr;   // D: N(col)=m
      #pragma unroll
      for (int v = 0; v < 8; v++) {
        const int nl = nsub * 16 + 8 * lh + v;   // D: M(row)=n
        float sv = s[v] > 0.f ? s[v] : 0.f;      // relu
        if (mg == kN - 1 && (n0s + nl) == kN - 1) sv = 0.f;  // att[-1,-1]=0
        ATT[msub * 16 + lr][nl] = (__bf16)sv;    // transpose into [m][n]
      }
    }
    __syncthreads();

    // ---- X_att accumulation: A = X (M=d,K=n), B = att (K=n,N=m) ----
    #pragma unroll
    for (int ds = 0; ds < 2; ds++) {
      v16bf a = frag_a(&XA[wv * 32 + ds * 16 + lr][0], 0, lh);
      #pragma unroll
      for (int msub = 0; msub < 4; msub++) {
        v16bf bb = frag_b(&ATT[msub * 16 + lr][0], 0, lh);
        acc[ds][msub] = wmma_bf16(a, bb, acc[ds][msub]);
      }
    }
    // ---- Y_att: threads 0..63 each own one att column m ----
    if (tid < kMT) {
      #pragma unroll
      for (int n = 0; n < kNT; n++)
        yacc += Yt[n] * (float)ATT[tid][n];
    }
  }

  // Write Xatt[b][m][d] bf16 (exactly the B-fragment layout k3 wants).
  #pragma unroll
  for (int ds = 0; ds < 2; ds++)
    #pragma unroll
    for (int msub = 0; msub < 4; msub++) {
      const int m  = m0 + msub * 16 + lr;          // D: N = m
      const int d0 = wv * 32 + ds * 16 + 8 * lh;   // D: M = d, 8 consecutive
      v8bf p;
      #pragma unroll
      for (int v = 0; v < 8; v++) p[v] = (__bf16)acc[ds][msub][v];
      *(v8bf*)(Xatt + ((size_t)b * kN + m) * kD + d0) = p;
    }
  // Y_out = Y + r * Y_att
  if (tid < kMT) {
    const float r = r_l[0];
    const int m = m0 + tid;
    out[((size_t)b * kRows + kD) * kN + m] = Yrow[m] + r * yacc;
  }
}

// ---------------------------------------------------------------------------
// k3: X_out = X + A_l * X_att   (WMMA, K=256), residual add fused.
// k outer / msub inner: one A-fragment load feeds 4 independent WMMA chains.
// ---------------------------------------------------------------------------
__global__ __launch_bounds__(256) void k3_out(
    const float* __restrict__ Z, const __bf16* __restrict__ Albf,
    const __bf16* __restrict__ Xatt, float* __restrict__ out) {
  const int b   = blockIdx.x / (kN / kMT);
  const int m0  = (blockIdx.x % (kN / kMT)) * kMT;
  const int tid = threadIdx.x;
  const int wv = tid >> 5, lane = tid & 31, lr = lane & 15, lh = lane >> 4;

  for (int ds = 0; ds < 2; ds++) {                  // 2 d-tiles per wave
    const int dt = wv * 2 + ds;
    const __bf16* arow = Albf + (size_t)(dt * 16 + lr) * kD;  // A_l row d, K=j
    const __bf16* bcol[4];
    #pragma unroll
    for (int msub = 0; msub < 4; msub++)
      bcol[msub] = Xatt + ((size_t)b * kN + m0 + msub * 16 + lr) * kD;

    v8f s[4] = {};                                  // 4 independent chains
    #pragma unroll
    for (int k = 0; k < kD; k += 32) {
      v16bf a = frag_a(arow, k, lh);
      #pragma unroll
      for (int msub = 0; msub < 4; msub++)
        s[msub] = wmma_bf16(a, frag_b(bcol[msub], k, lh), s[msub]);
    }
    #pragma unroll
    for (int msub = 0; msub < 4; msub++) {
      const int m = m0 + msub * 16 + lr;
      #pragma unroll
      for (int v = 0; v < 8; v++) {
        const int d = dt * 16 + 8 * lh + v;
        const size_t idx = ((size_t)b * kRows + d) * kN + m;  // X rows of Z/out
        out[idx] = Z[idx] + s[msub][v];
      }
    }
  }
}

// ---------------------------------------------------------------------------
extern "C" void kernel_launch(void* const* d_in, const int* in_sizes, int n_in,
                              void* d_out, int out_size, void* d_ws,
                              size_t ws_size, hipStream_t stream) {
  const float* Z   = (const float*)d_in[0];   // (8,257,2048)
  const float* A_l = (const float*)d_in[1];   // (256,256)
  const float* r_l = (const float*)d_in[2];   // (1,)
  const float* B_l = (const float*)d_in[3];   // (256,256)
  const float* C_l = (const float*)d_in[4];   // (256,256)
  float* out = (float*)d_out;

  char* ws = (char*)d_ws;
  const size_t wbytes = (size_t)kD * kD * sizeof(__bf16);        // 128 KB
  const size_t tbytes = (size_t)8 * kD * kN * sizeof(__bf16);    // 8 MB
  __bf16* Albf = (__bf16*)ws;                 ws += wbytes;
  __bf16* Blbf = (__bf16*)ws;                 ws += wbytes;
  __bf16* Clbf = (__bf16*)ws;                 ws += wbytes;
  __bf16* Xbf  = (__bf16*)ws;                 ws += tbytes;  // [b][d][n]
  __bf16* Ubf  = (__bf16*)ws;                 ws += tbytes;  // [b][n][d]
  __bf16* Wbf  = (__bf16*)ws;                 ws += tbytes;  // [b][m][d]
  __bf16* Xatt = (__bf16*)ws;                 ws += tbytes;  // [b][m][d]

  k0_convert<<<dim3((kD * kD + 255) / 256), dim3(256), 0, stream>>>(
      A_l, B_l, C_l, Albf, Blbf, Clbf);
  k1_uvx<<<dim3(8 * (kN / kNT1)), dim3(256), 0, stream>>>(
      Z, Blbf, Clbf, Xbf, Ubf, Wbf);
  k2_attn<<<dim3(8 * (kN / kMT)), dim3(256), 0, stream>>>(
      Z, Ubf, Wbf, Xbf, r_l, Xatt, out);
  k3_out<<<dim3(8 * (kN / kMT)), dim3(256), 0, stream>>>(
      Z, Albf, Xatt, out);
}
